// LSTM_8486855377039
// MI455X (gfx1250) — compile-verified, matrix operand order checked by reading, not run on hardware
//
#include <hip/hip_runtime.h>

// ---------------- Problem dims (fixed by the reference) ----------------
#define BB 256      // batch
#define TT 200      // timesteps
#define DD 512      // embedding dim
#define UU 1024     // LSTM units
#define GG 4096     // 4*U (gate width)

typedef __bf16 bf16_t;
typedef __attribute__((ext_vector_type(16))) __bf16 v16bf;
typedef __attribute__((ext_vector_type(8)))  __bf16 v8bf;
typedef __attribute__((ext_vector_type(8)))  float  v8f;
typedef __attribute__((ext_vector_type(4)))  unsigned int u32x4;
typedef __attribute__((ext_vector_type(8)))  int i32x8;
typedef __attribute__((ext_vector_type(4)))  int i32x4;

#if __has_builtin(__builtin_amdgcn_tensor_load_to_lds) && \
    __has_builtin(__builtin_amdgcn_s_wait_tensorcnt)
#define USE_TDM 1
#else
#define USE_TDM 0
#endif

// ---------------- Fragment loads ----------------
// 16x32 bf16 WMMA fragment (A-style striping) for this lane, from global.
// Per ISA 16-bit A layout: lanes 0-15 hold K {0..7, 16..23}, lanes 16-31 hold
// K {8..15, 24..31}; each half is a contiguous 16-byte chunk.
__device__ __forceinline__ v16bf load_frag(const bf16_t* __restrict__ base,
                                           int row, int ld, int k, int lane) {
  const bf16_t* p = base + (size_t)row * (size_t)ld + k + ((lane >> 4) << 3);
  v8bf lo = *reinterpret_cast<const v8bf*>(p);
  v8bf hi = *reinterpret_cast<const v8bf*>(p + 16);
  return __builtin_shufflevector(lo, hi, 0, 1, 2, 3, 4, 5, 6, 7,
                                 8, 9, 10, 11, 12, 13, 14, 15);
}

// Same fragment load but from an LDS-resident tile (emits ds_load_b128).
__device__ __forceinline__ v16bf load_frag_lds(const bf16_t* base,
                                               int row, int ld, int k, int lane) {
  const bf16_t* p = base + row * ld + k + ((lane >> 4) << 3);
  v8bf lo = *reinterpret_cast<const v8bf*>(p);
  v8bf hi = *reinterpret_cast<const v8bf*>(p + 16);
  return __builtin_shufflevector(lo, hi, 0, 1, 2, 3, 4, 5, 6, 7,
                                 8, 9, 10, 11, 12, 13, 14, 15);
}

__device__ __forceinline__ float sigmoidf(float x) {
  return 1.0f / (1.0f + __expf(-x));
}

// ---------------- TDM: 2-D tile (rows x cols bf16) global -> LDS ----------
// Issued by ONE wave (TDM issue ignores EXEC; one issue per wave = one DMA).
// Assumes the destination __shared__ array: lds byte address = low 32 bits of
// its flat address (LDS aperture maps addr[31:0] to LDS offset).
__device__ __forceinline__ void tdm_load_tile_bf16(const bf16_t* gsrc,
                                                   bf16_t* lds_dst,
                                                   unsigned rows,
                                                   unsigned cols,
                                                   unsigned tensor_rows,
                                                   unsigned stride_elems) {
#if USE_TDM
  unsigned long long ga = (unsigned long long)(const void*)gsrc;
  unsigned lds_off = (unsigned)(unsigned long long)(void*)lds_dst;
  u32x4 g0;
  g0[0] = 1u;                                     // count=1, is_restore=0, no gather
  g0[1] = lds_off;                                // LDS byte address
  g0[2] = (unsigned)(ga & 0xFFFFFFFFu);           // global_addr[31:0]
  g0[3] = (unsigned)((ga >> 32) & 0x01FFFFFFu)    // global_addr[56:32]
          | (2u << 30);                           // type = 2 ("image")
  unsigned long long s0 = stride_elems;           // tensor_dim0_stride (48b)
  i32x8 g1;
  g1[0] = (int)(1u << 16);                        // wg_mask=0, data_size=1 (2B)
  g1[1] = (int)((cols & 0xFFFFu) << 16);          // tensor_dim0[15:0]
  g1[2] = (int)(((cols >> 16) & 0xFFFFu)          // tensor_dim0[31:16]
          | ((tensor_rows & 0xFFFFu) << 16));     // tensor_dim1[15:0]
  g1[3] = (int)(((tensor_rows >> 16) & 0xFFFFu)   // tensor_dim1[31:16]
          | ((cols & 0xFFFFu) << 16));            // tile_dim0
  g1[4] = (int)(rows & 0xFFFFu);                  // tile_dim1 (tile_dim2 = 0)
  g1[5] = (int)(s0 & 0xFFFFFFFFull);              // dim0_stride[31:0]
  g1[6] = (int)((s0 >> 32) & 0xFFFFull);          // dim0_stride[47:32]
  g1[7] = 0;                                      // dim1_stride unused (2-D)
  i32x4 z4 = {0, 0, 0, 0};
  i32x8 z8 = {0, 0, 0, 0, 0, 0, 0, 0};
  // 6-arg form (this toolchain): (g0, g1, g2, g3, g4, cpol)
  __builtin_amdgcn_tensor_load_to_lds(g0, g1, z4, z4, z8, 0);
  __builtin_amdgcn_s_wait_tensorcnt(0);
#else
  // Cooperative fallback: tile spans full rows of the tensor only when
  // cols == stride_elems (always true for our call sites) -> contiguous copy.
  size_t total = (size_t)rows * cols;
  for (size_t i = (size_t)threadIdx.x * 8; i < total; i += (size_t)blockDim.x * 8)
    *reinterpret_cast<v8bf*>(lds_dst + i) =
        *reinterpret_cast<const v8bf*>(gsrc + i);
  (void)tensor_rows;
#endif
}

// ---------------- Utility kernels ----------------
__global__ void zero_u32(unsigned int* __restrict__ p, size_t n32) {
  size_t i = (size_t)blockIdx.x * blockDim.x + threadIdx.x;
  if (i < n32) p[i] = 0u;
}

// Wt[n*K + k] = (bf16) W[k*N + n]   (W is [K,N] fp32 row-major)
__global__ void transpose_cast(const float* __restrict__ W,
                               bf16_t* __restrict__ Wt, int K, int N) {
  size_t idx = (size_t)blockIdx.x * blockDim.x + threadIdx.x;
  size_t total = (size_t)K * (size_t)N;
  if (idx >= total) return;
  int k = (int)(idx % (size_t)K);
  int n = (int)(idx / (size_t)K);
  Wt[idx] = (bf16_t)W[(size_t)k * (size_t)N + n];
}

// x[t*B*D + b*D + d] = (bf16) emb[tokens[b*T + t]*D + d]   (time-major)
__global__ void gather_cast(const int* __restrict__ tokens,
                            const float* __restrict__ emb,
                            bf16_t* __restrict__ x) {
  size_t idx = (size_t)blockIdx.x * blockDim.x + threadIdx.x;
  size_t total = (size_t)TT * BB * DD;
  if (idx >= total) return;
  int d = (int)(idx % DD);
  size_t tb = idx / DD;
  int b = (int)(tb % BB);
  int t = (int)(tb / BB);
  int tok = tokens[(size_t)b * TT + t];
  x[idx] = (bf16_t)emb[(size_t)tok * DD + d];
}

// ---------------- Big parallel GEMM: C[M,N] = A[M,K] @ Bt[N,K]^T + bias ----
// A row-major bf16 (staged per-block into LDS via TDM), Bt row-major bf16
// (pre-transposed weights, streamed from L2), C bf16.
// Block = 8 waves sharing one 32-row A tile; each wave computes 32x64
// (2 M-subtiles x 4 N-subtiles = 8 WMMA accumulators).
__global__ void __launch_bounds__(256)
gemm_bias_bf16(const bf16_t* __restrict__ A,
               const bf16_t* __restrict__ Bt,
               const float* __restrict__ bias,
               bf16_t* __restrict__ C,
               int M, int N, int K) {
  __shared__ __align__(16) bf16_t shA[32 * 1024];   // 64 KB max (K <= 1024)
  const int lane = threadIdx.x & 31;
  const int wave = threadIdx.x >> 5;
  const int nGroups = N >> 9;                        // 512 cols per block
  const int mTile = blockIdx.x / nGroups;
  const int nGrp = blockIdx.x - mTile * nGroups;
  const int mBase = mTile * 32;
  const int nBase = (nGrp << 9) + (wave << 6);
  const int l15 = lane & 15;

  // Stage A tile [32, K] into LDS: one TDM transfer issued by wave 0.
#if USE_TDM
  if (wave == 0)
    tdm_load_tile_bf16(A + (size_t)mBase * K, shA, 32, K, M, K);
#else
  tdm_load_tile_bf16(A + (size_t)mBase * K, shA, 32, K, M, K);
#endif
  __syncthreads();

  v8f acc0 = {}, acc1 = {}, acc2 = {}, acc3 = {};
  v8f acc4 = {}, acc5 = {}, acc6 = {}, acc7 = {};
  for (int k = 0; k < K; k += 32) {
    v16bf a0 = load_frag_lds(shA, l15, K, k, lane);
    v16bf a1 = load_frag_lds(shA, 16 + l15, K, k, lane);
    v16bf b0 = load_frag(Bt, nBase + 0  + l15, K, k, lane);
    v16bf b1 = load_frag(Bt, nBase + 16 + l15, K, k, lane);
    v16bf b2 = load_frag(Bt, nBase + 32 + l15, K, k, lane);
    v16bf b3 = load_frag(Bt, nBase + 48 + l15, K, k, lane);
    acc0 = __builtin_amdgcn_wmma_f32_16x16x32_bf16(false, a0, false, b0,
                                                   (short)0, acc0, false, false);
    acc1 = __builtin_amdgcn_wmma_f32_16x16x32_bf16(false, a0, false, b1,
                                                   (short)0, acc1, false, false);
    acc2 = __builtin_amdgcn_wmma_f32_16x16x32_bf16(false, a0, false, b2,
                                                   (short)0, acc2, false, false);
    acc3 = __builtin_amdgcn_wmma_f32_16x16x32_bf16(false, a0, false, b3,
                                                   (short)0, acc3, false, false);
    acc4 = __builtin_amdgcn_wmma_f32_16x16x32_bf16(false, a1, false, b0,
                                                   (short)0, acc4, false, false);
    acc5 = __builtin_amdgcn_wmma_f32_16x16x32_bf16(false, a1, false, b1,
                                                   (short)0, acc5, false, false);
    acc6 = __builtin_amdgcn_wmma_f32_16x16x32_bf16(false, a1, false, b2,
                                                   (short)0, acc6, false, false);
    acc7 = __builtin_amdgcn_wmma_f32_16x16x32_bf16(false, a1, false, b3,
                                                   (short)0, acc7, false, false);
  }

  // C/D layout: VGPR j -> row rowBase + j + 8*(lane>=16), col colBase + l15
  const int mo = ((lane >> 4) << 3);
  const float bi0 = bias[nBase + 0  + l15];
  const float bi1 = bias[nBase + 16 + l15];
  const float bi2 = bias[nBase + 32 + l15];
  const float bi3 = bias[nBase + 48 + l15];
#pragma unroll
  for (int j = 0; j < 8; ++j) {
    size_t r0 = (size_t)(mBase + mo + j) * (size_t)N;
    size_t r1 = (size_t)(mBase + 16 + mo + j) * (size_t)N;
    C[r0 + nBase + 0  + l15] = (bf16_t)(acc0[j] + bi0);
    C[r0 + nBase + 16 + l15] = (bf16_t)(acc1[j] + bi1);
    C[r0 + nBase + 32 + l15] = (bf16_t)(acc2[j] + bi2);
    C[r0 + nBase + 48 + l15] = (bf16_t)(acc3[j] + bi3);
    C[r1 + nBase + 0  + l15] = (bf16_t)(acc4[j] + bi0);
    C[r1 + nBase + 16 + l15] = (bf16_t)(acc5[j] + bi1);
    C[r1 + nBase + 32 + l15] = (bf16_t)(acc6[j] + bi2);
    C[r1 + nBase + 48 + l15] = (bf16_t)(acc7[j] + bi3);
  }
}

// ---------------- Fused recurrent step --------------------------------------
// z = xz_t + h_in @ Ut^T ; gates i,f,g,o ; c = f*c + i*tanh(g); h = o*tanh(c)
// Block = 8 waves sharing one 16-row h_in tile (staged via TDM into LDS);
// each wave owns a 16x16 (batch,unit) patch + its 4 gate column blocks.
__global__ void __launch_bounds__(256)
lstm_step(const bf16_t* __restrict__ h_in,
          const bf16_t* __restrict__ Ut,
          const bf16_t* __restrict__ xz_t,
          float* __restrict__ c,
          bf16_t* __restrict__ h_out) {
  __shared__ __align__(16) bf16_t shH[16 * UU];     // 32 KB
  const int lane = threadIdx.x & 31;
  const int wave = threadIdx.x >> 5;
  const int uGroups = UU >> 7;                       // 8 (128 units per block)
  const int mTile = blockIdx.x / uGroups;
  const int uGrp = blockIdx.x - mTile * uGroups;
  const int mBase = mTile * 16;
  const int uBase = (uGrp << 7) + (wave << 4);
  const int l15 = lane & 15;

#if USE_TDM
  if (wave == 0)
    tdm_load_tile_bf16(h_in + (size_t)mBase * UU, shH, 16, UU, BB, UU);
#else
  tdm_load_tile_bf16(h_in + (size_t)mBase * UU, shH, 16, UU, BB, UU);
#endif
  __syncthreads();

  v8f accI = {}, accF = {}, accG = {}, accO = {};
  for (int k = 0; k < UU; k += 32) {
    v16bf a  = load_frag_lds(shH, l15, UU, k, lane);
    v16bf bi = load_frag(Ut, 0 * UU + uBase + l15, UU, k, lane);
    v16bf bf = load_frag(Ut, 1 * UU + uBase + l15, UU, k, lane);
    v16bf bg = load_frag(Ut, 2 * UU + uBase + l15, UU, k, lane);
    v16bf bo = load_frag(Ut, 3 * UU + uBase + l15, UU, k, lane);
    accI = __builtin_amdgcn_wmma_f32_16x16x32_bf16(false, a, false, bi,
                                                   (short)0, accI, false, false);
    accF = __builtin_amdgcn_wmma_f32_16x16x32_bf16(false, a, false, bf,
                                                   (short)0, accF, false, false);
    accG = __builtin_amdgcn_wmma_f32_16x16x32_bf16(false, a, false, bg,
                                                   (short)0, accG, false, false);
    accO = __builtin_amdgcn_wmma_f32_16x16x32_bf16(false, a, false, bo,
                                                   (short)0, accO, false, false);
  }

  const int mo = mBase + ((lane >> 4) << 3);
  const int u = uBase + l15;
#pragma unroll
  for (int j = 0; j < 8; ++j) {
    int b = mo + j;
    size_t zrow = (size_t)b * GG;
    float zi = accI[j] + (float)xz_t[zrow + 0 * UU + u];
    float zf = accF[j] + (float)xz_t[zrow + 1 * UU + u];
    float zg = accG[j] + (float)xz_t[zrow + 2 * UU + u];
    float zo = accO[j] + (float)xz_t[zrow + 3 * UU + u];
    size_t hc = (size_t)b * UU + u;
    float cn = sigmoidf(zf) * c[hc] + sigmoidf(zi) * tanhf(zg);
    c[hc] = cn;
    h_out[hc] = (bf16_t)(sigmoidf(zo) * tanhf(cn));
  }
}

// ---------------- Final dense + sigmoid -------------------------------------
__global__ void final_logits(const bf16_t* __restrict__ h,
                             const float* __restrict__ Wfc,
                             const float* __restrict__ bfc,
                             float* __restrict__ out) {
  __shared__ float red[256];
  int b = blockIdx.x;
  float s = 0.0f;
  for (int u = threadIdx.x; u < UU; u += blockDim.x)
    s += (float)h[(size_t)b * UU + u] * Wfc[u];
  red[threadIdx.x] = s;
  __syncthreads();
  for (int off = 128; off > 0; off >>= 1) {
    if ((int)threadIdx.x < off) red[threadIdx.x] += red[threadIdx.x + off];
    __syncthreads();
  }
  if (threadIdx.x == 0) out[b] = sigmoidf(red[0] + bfc[0]);
}

// ---------------- Host orchestration ----------------------------------------
extern "C" void kernel_launch(void* const* d_in, const int* in_sizes, int n_in,
                              void* d_out, int out_size, void* d_ws, size_t ws_size,
                              hipStream_t stream) {
  (void)in_sizes; (void)n_in; (void)out_size; (void)ws_size;

  const int*   tokens = (const int*)d_in[0];   // [B,T]
  const float* emb    = (const float*)d_in[1]; // [V,D]
  const float* W1     = (const float*)d_in[2]; // [D,4U]
  const float* U1     = (const float*)d_in[3]; // [U,4U]
  const float* b1     = (const float*)d_in[4]; // [4U]
  const float* W2     = (const float*)d_in[5]; // [U,4U]
  const float* U2     = (const float*)d_in[6]; // [U,4U]
  const float* b2     = (const float*)d_in[7]; // [4U]
  const float* Wfc    = (const float*)d_in[8]; // [U,1]
  const float* bfc    = (const float*)d_in[9]; // [1]
  float* out = (float*)d_out;                  // [B]

  // Workspace carve-up (256B aligned)
  char* ws = (char*)d_ws;
  size_t off = 0;
  auto carve = [&](size_t bytes) -> char* {
    char* p = ws + off;
    off = (off + bytes + 255) & ~(size_t)255;
    return p;
  };
  bf16_t* x_bf   = (bf16_t*)carve((size_t)TT * BB * DD * 2); // embedded input, time-major
  bf16_t* h_seq  = (bf16_t*)carve((size_t)TT * BB * UU * 2); // layer-1 hidden sequence
  bf16_t* xz     = (bf16_t*)carve((size_t)TT * BB * GG * 2); // input projections (reused L1/L2)
  bf16_t* W1t    = (bf16_t*)carve((size_t)GG * DD * 2);
  bf16_t* U1t    = (bf16_t*)carve((size_t)GG * UU * 2);
  bf16_t* W2t    = (bf16_t*)carve((size_t)GG * UU * 2);
  bf16_t* U2t    = (bf16_t*)carve((size_t)GG * UU * 2);
  bf16_t* h_zero = (bf16_t*)carve((size_t)BB * UU * 2);
  bf16_t* h_a    = (bf16_t*)carve((size_t)BB * UU * 2);
  bf16_t* h_b    = (bf16_t*)carve((size_t)BB * UU * 2);
  float*  c_buf  = (float*)carve((size_t)BB * UU * 4);

  const int TPB = 256;
  auto blocks = [](size_t n, int tpb) { return (unsigned)((n + tpb - 1) / tpb); };

  // Zero h0 and c
  zero_u32<<<blocks((size_t)BB * UU / 2, TPB), TPB, 0, stream>>>(
      (unsigned int*)h_zero, (size_t)BB * UU / 2);
  zero_u32<<<blocks((size_t)BB * UU, TPB), TPB, 0, stream>>>(
      (unsigned int*)c_buf, (size_t)BB * UU);

  // Transpose+cast weights: [K,4U] fp32 -> [4U,K] bf16
  transpose_cast<<<blocks((size_t)DD * GG, TPB), TPB, 0, stream>>>(W1, W1t, DD, GG);
  transpose_cast<<<blocks((size_t)UU * GG, TPB), TPB, 0, stream>>>(U1, U1t, UU, GG);
  transpose_cast<<<blocks((size_t)UU * GG, TPB), TPB, 0, stream>>>(W2, W2t, UU, GG);
  transpose_cast<<<blocks((size_t)UU * GG, TPB), TPB, 0, stream>>>(U2, U2t, UU, GG);

  // Embedding gather -> bf16 (time-major [T,B,D])
  gather_cast<<<blocks((size_t)TT * BB * DD, TPB), TPB, 0, stream>>>(tokens, emb, x_bf);

  const int M = TT * BB;                                    // 51200 rows
  const unsigned gemmBlocks = (unsigned)((M / 32) * (GG / 512)); // 12800

  // Layer 1: input projections for all timesteps in one parallel GEMM
  gemm_bias_bf16<<<gemmBlocks, TPB, 0, stream>>>(x_bf, W1t, b1, xz, M, GG, DD);

  // Layer 1 recurrence: h_out(t) lives directly in h_seq
  const unsigned stepBlocks = (unsigned)((BB / 16) * (UU / 128)); // 128
  for (int t = 0; t < TT; ++t) {
    const bf16_t* hin = (t == 0) ? h_zero : (h_seq + (size_t)(t - 1) * BB * UU);
    bf16_t* hout = h_seq + (size_t)t * BB * UU;
    lstm_step<<<stepBlocks, TPB, 0, stream>>>(
        hin, U1t, xz + (size_t)t * BB * GG, c_buf, hout);
  }

  // Layer 2: input projections from h_seq (reuse xz buffer)
  gemm_bias_bf16<<<gemmBlocks, TPB, 0, stream>>>(h_seq, W2t, b2, xz, M, GG, UU);

  // Reset cell state, run layer-2 recurrence with ping-pong h buffers
  zero_u32<<<blocks((size_t)BB * UU, TPB), TPB, 0, stream>>>(
      (unsigned int*)c_buf, (size_t)BB * UU);
  bf16_t* hb[2] = {h_a, h_b};
  for (int t = 0; t < TT; ++t) {
    const bf16_t* hin = (t == 0) ? h_zero : hb[(t - 1) & 1];
    lstm_step<<<stepBlocks, TPB, 0, stream>>>(
        hin, U2t, xz + (size_t)t * BB * GG, c_buf, hb[t & 1]);
  }

  // Final dense + sigmoid -> [B]
  final_logits<<<BB, TPB, 0, stream>>>(hb[(TT - 1) & 1], Wfc, bfc, out);
}
